// ClassSpecificResNet_38199439130878
// MI455X (gfx1250) — compile-verified, matrix operand order checked by reading, not run on hardware
//
#include <hip/hip_runtime.h>

// ---------------------------------------------------------------------------
// CDNA5 (gfx1250, wave32) implementation.
// All GEMM-shaped work (implicit grouped conv in NHWC, qkv 1x1, Q*K^T,
// attn*V via v^T) runs through V_WMMA_F32_16X16X32_F16 with pure NT
// addressing: every fragment is two unconditional global_load_b128s.
// One 16x16 tile per wave, 4 waves (128 threads) per block.
// ---------------------------------------------------------------------------

typedef _Float16 half_t;
typedef __attribute__((ext_vector_type(16))) _Float16 v16h;
typedef __attribute__((ext_vector_type(8)))  _Float16 h8;
typedef __attribute__((ext_vector_type(8)))  float    v8f;

#define NCLS 10
#define NHEADS 4

// Load one 16x32 fp16 fragment for this lane from a K-contiguous row.
// p must point at  row_base + k0 + hi*8  (8-half aligned).
// Halves j=0..7  <-> K = k0 + hi*8 + j
// Halves j=8..15 <-> K = k0 + 16 + hi*8 + (j-8)     (CDNA5 ISA 7.12.2)
__device__ __forceinline__ v16h frag_nt(const half_t* p) {
  h8 a = *(const h8*)(p);
  h8 b = *(const h8*)(p + 16);
  return __builtin_shufflevector(a, b, 0, 1, 2, 3, 4, 5, 6, 7,
                                 8, 9, 10, 11, 12, 13, 14, 15);
}

// ---------------------------------------------------------------------------
// Batched NT WMMA GEMM:  C[z] = alpha * A[z] * B[z]^T  (+ class-mask epilogue)
// A: [M][K] lda, B: [N][K] ldb, both K-contiguous.  M,N % 16 == 0, K % 32 == 0.
// z decomposes as zb = z/HB, zh = z%HB with separate strides (addresses q/k/v
// slices of the packed qkv tensor directly).
// ycls != nullptr: columns outside [y*allocw, y*allocw+allocw) -> -inf.
// ---------------------------------------------------------------------------
__global__ void gemm_wmma_nt(const half_t* __restrict__ A,
                             const half_t* __restrict__ B,
                             float* __restrict__ C,
                             int M, int N, int K,
                             int lda, int ldb, int ldc,
                             long sAb, long sAh, long sBb, long sBh,
                             long sCb, long sCh,
                             int HB, float alpha,
                             const int* __restrict__ ycls, int allocw) {
  int lane = threadIdx.x & 31;
  int wave = threadIdx.x >> 5;
  int hi = lane >> 4, lo = lane & 15;
  int m0 = (blockIdx.x * 4 + wave) * 16;
  int n0 = blockIdx.y * 16;
  if (m0 >= M) return;
  int z = blockIdx.z;
  int zb = z / HB, zh = z % HB;
  const half_t* pa = A + zb * sAb + zh * sAh + (long)(m0 + lo) * lda + hi * 8;
  const half_t* pb = B + zb * sBb + zh * sBh + (long)(n0 + lo) * ldb + hi * 8;
  float* Cz = C + zb * sCb + zh * sCh;

  v8f acc = {};
  for (int k0 = 0; k0 < K; k0 += 32) {
    v16h a = frag_nt(pa + k0);
    v16h b = frag_nt(pb + k0);
    acc = __builtin_amdgcn_wmma_f32_16x16x32_f16(
        false, a, false, b, (short)0, acc, false, false);
  }
  int n = n0 + lo;
  int jlo = 0, jhi = 0x7fffffff;
  if (ycls) { int cls = ycls[zb]; jlo = cls * allocw; jhi = jlo + allocw; }
#pragma unroll
  for (int r = 0; r < 8; ++r) {
    int m = m0 + r + hi * 8;
    float v = acc[r] * alpha;
    if (ycls && !(n >= jlo && n < jhi)) v = -__builtin_inff();
    Cz[(long)m * ldc + n] = v;
  }
}

// ---------------------------------------------------------------------------
// Implicit-GEMM grouped conv, NHWC activations, weights pre-permuted to
// [Cout][(kh*ks+kw)*Cig + ci] with Cig % 32 == 0 (stem channel-padded to 32).
// Each 32-wide K-step sits inside one (kh,kw) tap -> A fragment = two
// unconditional b128 loads from a clamped address + one vector select for
// zero padding; W fragment = two unconditional b128 loads.
// Output f32 and/or f16, NCHW.  M = B*Ho*Wo (% 16 == 0), Cog % 16 == 0.
// ---------------------------------------------------------------------------
__global__ void conv_wmma_nhwc(const half_t* __restrict__ X,
                               const half_t* __restrict__ Wt,
                               const float* __restrict__ bias,
                               float* __restrict__ outF,
                               half_t* __restrict__ outH,
                               int Bn, int Cin, int Hin, int Win, int Cig,
                               int Cout, int Ho, int Wo,
                               int ks, int stride, int pad, int groups) {
  int lane = threadIdx.x & 31;
  int wave = threadIdx.x >> 5;
  int hi = lane >> 4, lo = lane & 15;
  int HWo = Ho * Wo;
  int M = Bn * HWo;
  int m0 = (blockIdx.x * 4 + wave) * 16;
  if (m0 >= M) return;
  int Cog = Cout / groups;
  int tilesN = Cog >> 4;
  int g = blockIdx.y / tilesN;
  int n0 = (blockIdx.y % tilesN) * 16;
  int K = ks * ks * Cig;

  int rA = m0 + lo;
  int b = rA / HWo;
  int rem = rA - b * HWo;
  int oh = rem / Wo, ow = rem - (rem / Wo) * Wo;
  int ihBase = oh * stride - pad;
  int iwBase = ow * stride - pad;

  const half_t* Xb = X + ((long)b * Hin * Win) * Cin + g * Cig + hi * 8;
  const half_t* Wg = Wt + (long)(g * Cog + n0 + lo) * K + hi * 8;

  v8f acc = {};
  int ksq = ks * ks;
  for (int s = 0; s < ksq; ++s) {
    int kh = s / ks, kw = s - (s / ks) * ks;
    int ih = ihBase + kh, iw = iwBase + kw;
    bool valid = ((unsigned)ih < (unsigned)Hin) && ((unsigned)iw < (unsigned)Win);
    int ihc = ih < 0 ? 0 : (ih >= Hin ? Hin - 1 : ih);
    int iwc = iw < 0 ? 0 : (iw >= Win ? Win - 1 : iw);
    const half_t* xp = Xb + ((long)ihc * Win + iwc) * Cin;
    const half_t* wp = Wg + (long)s * Cig;
    for (int cb = 0; cb < Cig; cb += 32) {
      v16h av = frag_nt(xp + cb);
      if (!valid) av = (v16h){};
      v16h bv = frag_nt(wp + cb);
      acc = __builtin_amdgcn_wmma_f32_16x16x32_f16(
          false, av, false, bv, (short)0, acc, false, false);
    }
  }

  int oc = g * Cog + n0 + lo;
  float bv = bias ? bias[oc] : 0.0f;
#pragma unroll
  for (int r = 0; r < 8; ++r) {
    int m = m0 + r + hi * 8;
    int b2 = m / HWo;
    int r2 = m - b2 * HWo;
    long oidx = ((long)b2 * Cout + oc) * HWo + r2;
    float v = acc[r] + bv;
    if (outF) outF[oidx] = v;
    if (outH) outH[oidx] = (half_t)v;
  }
}

// ---------------------------------------------------------------------------
// Pack input x (f32 NCHW, C=3) -> f16 NHWC with C padded to 32.
// ---------------------------------------------------------------------------
__global__ void pack_x_kernel(const float* __restrict__ x, half_t* __restrict__ dst,
                              long total) {
  long t = (long)blockIdx.x * blockDim.x + threadIdx.x;
  if (t >= total) return;
  int c = (int)(t & 31);
  long r = t >> 5;
  int sp = (int)(r & 1023);
  int b = (int)(r >> 10);
  half_t v = (half_t)0.0f;
  if (c < 3) v = (half_t)x[((long)b * 3 + c) * 1024 + sp];
  dst[t] = v;
}

// ---------------------------------------------------------------------------
// Weight pad+permute: src [Cout][CigS][ks][ks] f32 ->
//                     dst [Cout][(kh*ks+kw)*CigD + ci] f16 (ci >= CigS -> 0).
// ---------------------------------------------------------------------------
__global__ void wpad_kernel(const float* __restrict__ src, half_t* __restrict__ dst,
                            int CigS, int CigD, int ks, long total) {
  long t = (long)blockIdx.x * blockDim.x + threadIdx.x;
  if (t >= total) return;
  int ci = (int)(t % CigD);
  long u = t / CigD;
  int ksq = ks * ks;
  int tap = (int)(u % ksq);
  int oc = (int)(u / ksq);
  int kh = tap / ks, kw = tap - (tap / ks) * ks;
  half_t v = (half_t)0.0f;
  if (ci < CigS) v = (half_t)src[(((long)oc * CigS + ci) * ks + kh) * ks + kw];
  dst[t] = v;
}

// ---------------------------------------------------------------------------
// v^T: qkv (NCHW f16, v slice) -> vT[z=(b,h)][sp][i], K=i contiguous for NT.
// ---------------------------------------------------------------------------
__global__ void vtrans_kernel(const half_t* __restrict__ qkv, half_t* __restrict__ vT,
                              int Ctot, int d, int HW, long total) {
  long t = (long)blockIdx.x * blockDim.x + threadIdx.x;
  if (t >= total) return;
  int i = (int)(t % d);
  long r = t / d;
  int sp = (int)(r % HW);
  int z = (int)(r / HW);
  int b = z / NHEADS, h = z - (z / NHEADS) * NHEADS;
  vT[t] = qkv[((long)b * 3 * Ctot + 2 * Ctot + h * d + i) * HW + sp];
}

// ---------------------------------------------------------------------------
// BatchNorm batch statistics over (B, HW) per channel (NCHW f32).
// ---------------------------------------------------------------------------
__global__ void bn_stats_kernel(const float* __restrict__ x,
                                float* __restrict__ mean, float* __restrict__ var,
                                int Bn, int C, int HW) {
  int c = blockIdx.x;
  int n = Bn * HW;
  float s = 0.f, s2 = 0.f;
  for (int i = threadIdx.x; i < n; i += blockDim.x) {
    int b = i / HW, r = i - b * HW;
    float v = x[((long)b * C + c) * HW + r];
    s += v; s2 += v * v;
  }
  __shared__ float sh[256], sh2[256];
  sh[threadIdx.x] = s; sh2[threadIdx.x] = s2;
  __syncthreads();
  for (int st = blockDim.x >> 1; st > 0; st >>= 1) {
    if ((int)threadIdx.x < st) {
      sh[threadIdx.x] += sh[threadIdx.x + st];
      sh2[threadIdx.x] += sh2[threadIdx.x + st];
    }
    __syncthreads();
  }
  if (threadIdx.x == 0) {
    float m = sh[0] / n;
    mean[c] = m;
    var[c] = sh2[0] / n - m * m;
  }
}

// ---------------------------------------------------------------------------
// y = [relu]( bn(x)*g + b  [+ res] ); f32 out NCHW, f16 out NHWC (for convs).
// ---------------------------------------------------------------------------
__global__ void bn_act_kernel(const float* __restrict__ x,
                              const float* __restrict__ mean, const float* __restrict__ var,
                              const float* __restrict__ gamma, const float* __restrict__ beta,
                              const float* __restrict__ res,
                              float* __restrict__ outF, half_t* __restrict__ outH,
                              int C, int HW, long total, int relu) {
  long i = (long)blockIdx.x * blockDim.x + threadIdx.x;
  if (i >= total) return;
  int hw = (int)(i % HW);
  int c = (int)((i / HW) % C);
  int b = (int)(i / ((long)HW * C));
  float v = (x[i] - mean[c]) * rsqrtf(var[c] + 1e-5f) * gamma[c] + beta[c];
  if (res) v += res[i];
  if (relu) v = fmaxf(v, 0.f);
  if (outF) outF[i] = v;
  if (outH) outH[((long)b * HW + hw) * C + c] = (half_t)v;
}

// ---------------------------------------------------------------------------
// Row softmax, in-place f32 + f16 copy.  One block per row (rowlen <= 320).
// ---------------------------------------------------------------------------
__global__ void softmax_kernel(float* __restrict__ dots, half_t* __restrict__ outH,
                               int rowlen) {
  long row = blockIdx.x;
  float* p = dots + row * rowlen;
  __shared__ float sh[256];
  int t = threadIdx.x;
  float mx = -__builtin_inff();
  for (int j = t; j < rowlen; j += blockDim.x) mx = fmaxf(mx, p[j]);
  sh[t] = mx; __syncthreads();
  for (int st = blockDim.x >> 1; st > 0; st >>= 1) {
    if (t < st) sh[t] = fmaxf(sh[t], sh[t + st]);
    __syncthreads();
  }
  mx = sh[0]; __syncthreads();
  float s = 0.f;
  for (int j = t; j < rowlen; j += blockDim.x) s += expf(p[j] - mx);
  sh[t] = s; __syncthreads();
  for (int st = blockDim.x >> 1; st > 0; st >>= 1) {
    if (t < st) sh[t] += sh[t + st];
    __syncthreads();
  }
  float inv = 1.f / sh[0];
  for (int j = t; j < rowlen; j += blockDim.x) {
    float e = expf(p[j] - mx) * inv;
    p[j] = e;
    outH[row * rowlen + j] = (half_t)e;
  }
}

// ---------------------------------------------------------------------------
// 4x4 average pool of (B,C,16,16) -> (B, C*16), reference .reshape(B,-1) order.
// ---------------------------------------------------------------------------
__global__ void pool_kernel(const float* __restrict__ x, float* __restrict__ out,
                            int Bn, int C) {
  long idx = (long)blockIdx.x * blockDim.x + threadIdx.x;
  long total = (long)Bn * C * 16;
  if (idx >= total) return;
  int pj = (int)(idx & 3);
  long t1 = idx >> 2;
  int pi = (int)(t1 & 3);
  long t2 = t1 >> 2;
  int c = (int)(t2 % C);
  int b = (int)(t2 / C);
  const float* base = x + (((long)b * C + c) * 16 + pi * 4) * 16 + pj * 4;
  float s = 0.f;
#pragma unroll
  for (int a = 0; a < 4; ++a)
#pragma unroll
    for (int bb = 0; bb < 4; ++bb) s += base[a * 16 + bb];
  out[((long)b * C + c) * 16 + pi * 4 + pj] = s * (1.f / 16.f);
}

// ---------------------------------------------------------------------------
// Tiny FC: logits[b,n] = feat[b] . fc_w[n] + fc_b[n]   (32x10, K=20480)
// ---------------------------------------------------------------------------
__global__ void fc_kernel(const float* __restrict__ feat, const float* __restrict__ w,
                          const float* __restrict__ bias, float* __restrict__ logits,
                          int Bn, int NCo, int K) {
  int i = blockIdx.x * blockDim.x + threadIdx.x;
  if (i >= Bn * NCo) return;
  int b = i / NCo, n = i - (i / NCo) * NCo;
  const float* f = feat + (long)b * K;
  const float* ww = w + (long)n * K;
  float s = bias[n];
  for (int k = 0; k < K; ++k) s += f[k] * ww[k];
  logits[(long)b * NCo + n] = s;
}

// ---------------------------------------------------------------------------
// Host orchestration
// ---------------------------------------------------------------------------
extern "C" void kernel_launch(void* const* d_in, const int* in_sizes, int n_in,
                              void* d_out, int out_size, void* d_ws, size_t ws_size,
                              hipStream_t stream) {
  (void)in_sizes; (void)n_in; (void)out_size; (void)ws_size;
  const int Bn = 32;
  const int cinA[4]  = {640, 640, 640, 1280};
  const int coutA[4] = {640, 640, 1280, 1280};
  const int strA[4]  = {1, 1, 2, 1};
  const int dnA[4]   = {0, 0, 1, 0};

  const float* x_in   = (const float*)d_in[0];
  const int*   ycls   = (const int*)d_in[1];
  const float* stem_w = (const float*)d_in[2];
  const float* stem_g = (const float*)d_in[3];
  const float* stem_b = (const float*)d_in[4];
  struct BP { const float *c1w,*c1b,*ag,*ab,*qkvw,*b1g,*b1b,*dw,*dg,*db; } bp[4];
  int pi = 5;
  for (int i = 0; i < 4; ++i) {
    bp[i].c1w  = (const float*)d_in[pi++];
    bp[i].c1b  = (const float*)d_in[pi++];
    bp[i].ag   = (const float*)d_in[pi++];
    bp[i].ab   = (const float*)d_in[pi++];
    bp[i].qkvw = (const float*)d_in[pi++];
    bp[i].b1g  = (const float*)d_in[pi++];
    bp[i].b1b  = (const float*)d_in[pi++];
    if (dnA[i]) {
      bp[i].dw = (const float*)d_in[pi++];
      bp[i].dg = (const float*)d_in[pi++];
      bp[i].db = (const float*)d_in[pi++];
    } else { bp[i].dw = bp[i].dg = bp[i].db = nullptr; }
  }
  const float* fc_w = (const float*)d_in[pi++];
  const float* fc_b = (const float*)d_in[pi++];
  float* out = (float*)d_out;

  char* wsb = (char*)d_ws;
  size_t off = 0;
  auto wsAlloc = [&](size_t bytes) -> void* {
    void* p = wsb + off;
    off += (bytes + 255) & ~(size_t)255;
    return p;
  };
  const long ACTMAX = 32L * 640 * 32 * 32;            // 20.97M elems
  float*  BUF_X    = (float*)wsAlloc(ACTMAX * 4);     // block input/residual, NCHW
  half_t* BUF_XH   = (half_t*)wsAlloc(ACTMAX * 2);    // NHWC f16 for convs
  float*  BUF_C1   = (float*)wsAlloc(ACTMAX * 4);     // conv1 out -> o, NCHW
  half_t* BUF_C1H  = (half_t*)wsAlloc(ACTMAX * 2);    // NHWC f16 (qkv input)
  half_t* BUF_QKVH = (half_t*)wsAlloc(32L * 1920 * 1024 * 2);  // NCHW f16
  half_t* BUF_VT   = (half_t*)wsAlloc(32L * 640 * 1024 * 2);   // v^T per (b,h)
  half_t* BUF_ATTNH= (half_t*)wsAlloc(32L * 4 * 320 * 320 * 2);
  float*  BUF_SC   = (float*)wsAlloc(32L * 1280 * 16 * 16 * 4);
  float*  MEAN1 = (float*)wsAlloc(1280 * 4);
  float*  VAR1  = (float*)wsAlloc(1280 * 4);
  float*  MEAN2 = (float*)wsAlloc(1280 * 4);
  float*  VAR2  = (float*)wsAlloc(1280 * 4);
  float*  MEAND = (float*)wsAlloc(1280 * 4);
  float*  VARD  = (float*)wsAlloc(1280 * 4);
  half_t* XINH  = (half_t*)wsAlloc(32L * 1024 * 32 * 2);  // NHWC, C padded to 32
  float*  POOL  = (float*)wsAlloc(32L * 20480 * 4);
  half_t* whStem = (half_t*)wsAlloc(640L * 9 * 32 * 2);   // Cig padded 3 -> 32
  half_t* whC1[4]; half_t* whQkv[4];
  const long c1sz[4]  = {368640, 368640, 737280, 1474560};
  const long qkvsz[4] = {122880, 122880, 245760, 491520};
  for (int i = 0; i < 4; ++i) whC1[i]  = (half_t*)wsAlloc(c1sz[i] * 2);
  for (int i = 0; i < 4; ++i) whQkv[i] = (half_t*)wsAlloc(qkvsz[i] * 2);
  half_t* whDown = (half_t*)wsAlloc(81920L * 2);

  auto wpadL = [&](const float* s, half_t* d, int Cout, int CigS, int CigD, int ks) {
    long total = (long)Cout * ks * ks * CigD;
    wpad_kernel<<<dim3((unsigned)((total + 255) / 256)), dim3(256), 0, stream>>>(
        s, d, CigS, CigD, ks, total);
  };
  // ---- pack/pad input + weights ----
  {
    long total = 32L * 1024 * 32;
    pack_x_kernel<<<dim3((unsigned)((total + 255) / 256)), dim3(256), 0, stream>>>(
        x_in, XINH, total);
  }
  wpadL(stem_w, whStem, 640, 3, 32, 3);
  for (int i = 0; i < 4; ++i) {
    int cig = cinA[i] / NCLS;
    wpadL(bp[i].c1w, whC1[i], coutA[i], cig, cig, 3);
    int cq = coutA[i] / NCLS;
    wpadL(bp[i].qkvw, whQkv[i], 3 * coutA[i], cq, cq, 1);
  }
  wpadL(bp[2].dw, whDown, 1280, 64, 64, 1);

  auto convL = [&](const half_t* X, const half_t* W, const float* bias,
                   float* oF, half_t* oH, int Cin, int Cig, int Hin,
                   int Cout, int Ho, int ks, int stride, int pad, int groups) {
    int M = Bn * Ho * Ho;
    int Cog = Cout / groups;
    dim3 grid((unsigned)((M + 63) / 64), (unsigned)(groups * (Cog / 16)), 1);
    conv_wmma_nhwc<<<grid, dim3(128), 0, stream>>>(
        X, W, bias, oF, oH, Bn, Cin, Hin, Hin, Cig, Cout, Ho, Ho,
        ks, stride, pad, groups);
  };
  auto statsL = [&](const float* x, float* m, float* v, int C, int HW) {
    bn_stats_kernel<<<dim3((unsigned)C), dim3(256), 0, stream>>>(x, m, v, Bn, C, HW);
  };
  auto bnL = [&](const float* x, const float* m, const float* v,
                 const float* g, const float* b, const float* res,
                 float* oF, half_t* oH, int C, int HW, int relu) {
    long total = (long)Bn * C * HW;
    bn_act_kernel<<<dim3((unsigned)((total + 255) / 256)), dim3(256), 0, stream>>>(
        x, m, v, g, b, res, oF, oH, C, HW, total, relu);
  };

  // ---- stem: conv 3x3 (padded Cin=32, groups=1) -> BN -> ReLU ----
  convL(XINH, whStem, nullptr, BUF_C1, nullptr, 32, 32, 32, 640, 32, 3, 1, 1, 1);
  statsL(BUF_C1, MEAN1, VAR1, 640, 1024);
  bnL(BUF_C1, MEAN1, VAR1, stem_g, stem_b, nullptr, BUF_X, BUF_XH, 640, 1024, 1);

  long attnOff[4];
  long o = 320;
  for (int i = 0; i < 4; ++i) {
    int d = coutA[i] / NHEADS;
    attnOff[i] = o;
    o += (long)Bn * NHEADS * d * d;
  }

  int hin = 32;
  for (int blk = 0; blk < 4; ++blk) {
    const int cin = cinA[blk], cout = coutA[blk], stride = strA[blk];
    const int Ho = hin / stride;
    const int HW = Ho * Ho;
    const int d  = cout / NHEADS;
    const int allocw = cout / (NCLS * NHEADS);
    const float scale = 1.0f / sqrtf((float)cout);

    // conv1 (grouped 3x3) + bias -> BUF_C1 (f32 NCHW)
    convL(BUF_XH, whC1[blk], bp[blk].c1b, BUF_C1, nullptr,
          cin, cin / NCLS, hin, cout, Ho, 3, stride, 1, NCLS);
    // abn + relu -> BUF_C1H (f16 NHWC) only (f32 not needed downstream)
    statsL(BUF_C1, MEAN1, VAR1, cout, HW);
    bnL(BUF_C1, MEAN1, VAR1, bp[blk].ag, bp[blk].ab, nullptr,
        nullptr, BUF_C1H, cout, HW, 1);
    // qkv (grouped 1x1) -> BUF_QKVH (f16 NCHW)
    convL(BUF_C1H, whQkv[blk], nullptr, nullptr, BUF_QKVH,
          cout, cout / NCLS, Ho, 3 * cout, Ho, 1, 1, 0, NCLS);
    // v^T for the attn*V NT GEMM
    {
      long total = (long)Bn * cout * HW;
      vtrans_kernel<<<dim3((unsigned)((total + 255) / 256)), dim3(256), 0, stream>>>(
          BUF_QKVH, BUF_VT, cout, d, HW, total);
    }

    float* dots = out + attnOff[blk];
    // dots = q @ k^T * scale + class mask  -> d_out attn slab (f32)
    {
      dim3 grid((unsigned)((d + 63) / 64), (unsigned)(d / 16), (unsigned)(Bn * NHEADS));
      gemm_wmma_nt<<<grid, dim3(128), 0, stream>>>(
          BUF_QKVH,                      // q
          BUF_QKVH + (long)cout * HW,    // k
          dots,
          d, d, HW, HW, HW, d,
          3L * cout * HW, (long)d * HW,
          3L * cout * HW, (long)d * HW,
          (long)NHEADS * d * d, (long)d * d,
          NHEADS, scale, ycls, allocw);
    }
    // softmax rows, in place + f16 copy
    softmax_kernel<<<dim3((unsigned)(Bn * NHEADS * d)), dim3(256), 0, stream>>>(
        dots, BUF_ATTNH, d);
    // o = attn @ vT^T -> BUF_C1 (f32, NCHW with channel = h*d + i)
    {
      dim3 grid((unsigned)((d + 63) / 64), (unsigned)(HW / 16), (unsigned)(Bn * NHEADS));
      gemm_wmma_nt<<<grid, dim3(128), 0, stream>>>(
          BUF_ATTNH, BUF_VT, BUF_C1,
          d, HW, d, d, d, HW,
          (long)NHEADS * d * d, (long)d * d,
          (long)NHEADS * HW * d, (long)HW * d,
          (long)cout * HW, (long)d * HW,
          NHEADS, 1.0f, nullptr, 0);
    }

    // bn1 + residual + relu -> BUF_X (new x, NCHW) + BUF_XH (NHWC)
    statsL(BUF_C1, MEAN2, VAR2, cout, HW);
    if (dnA[blk]) {
      convL(BUF_XH, whDown, nullptr, BUF_SC, nullptr,
            cin, cin / NCLS, hin, cout, Ho, 1, stride, 0, NCLS);
      statsL(BUF_SC, MEAND, VARD, cout, HW);
      bnL(BUF_SC, MEAND, VARD, bp[blk].dg, bp[blk].db, nullptr,
          BUF_SC, nullptr, cout, HW, 0);
      bnL(BUF_C1, MEAN2, VAR2, bp[blk].b1g, bp[blk].b1b, BUF_SC,
          BUF_X, BUF_XH, cout, HW, 1);
    } else {
      bnL(BUF_C1, MEAN2, VAR2, bp[blk].b1g, bp[blk].b1b, BUF_X,
          BUF_X, BUF_XH, cout, HW, 1);
    }
    hin = Ho;
  }

  // ---- 4x4 average pool + FC ----
  {
    long total = (long)Bn * 1280 * 16;
    pool_kernel<<<dim3((unsigned)((total + 255) / 256)), dim3(256), 0, stream>>>(
        BUF_X, POOL, Bn, 1280);
    fc_kernel<<<dim3((unsigned)((Bn * NCLS + 63) / 64)), dim3(64), 0, stream>>>(
        POOL, fc_w, fc_b, out, Bn, NCLS, 1280 * 16);
  }
}
